// Embedding_53850299958008
// MI455X (gfx1250) — matrix operand order repaired and checked by reading, not compiled, not run on hardware
//
#include <hip/hip_runtime.h>

// ---------------------------------------------------------------------------
// Fused PointNet-style embedding for MI455X (gfx1250, wave32, WMMA).
// All BN affines folded into weights; heavy GEMMs via v_wmma_f32_16x16x32_f16.
// k-max pooling done by LDS transpose (no bpermute chains).
// ---------------------------------------------------------------------------

#define EPSV 1e-5f

constexpr int CIN   = 8;
constexpr int COUT  = 128;
constexpr int NPTS  = 65536;
constexpr int KNBR  = 16;      // neighbors array K (uses k=1..15)
constexpr int NCOL  = 16;      // columns per workgroup
constexpr int NQ    = 256;     // NCOL * 16 neighbor slots
constexpr int SH_STRIDE  = 136; // fp16 elems per q-column of H (16B aligned, bank-spread)
constexpr int CAT_STRIDE = 264; // fp16 elems per column of cat (16B aligned, bank-spread)
constexpr int SRED_ROW  = 20;   // f32 per k-row of reduction scratch (80B: aligned, bank-tiling)
constexpr int SRED_TILE = 336;  // f32 per 16x16 tile (320 + pad 16 -> halves hit disjoint banks)

typedef __attribute__((ext_vector_type(16))) _Float16 v16h;
typedef __attribute__((ext_vector_type(8)))  _Float16 v8h;
typedef __attribute__((ext_vector_type(8)))  float    v8f;

union V16U { v16h v; v8h h[2]; };

// ---------------------------------------------------------------------------
// Setup: fold BN1/BN2/BN3 into W1', bias1, W2', bias2; convert wB, wf to fp16.
//   point_emb = W1' @ x + bias1            (W1'[c][o] layout, c-major)
//   H         = relu(W2' @ (x_g - x_c) + bias2)
// ---------------------------------------------------------------------------
__global__ __launch_bounds__(128) void setup_kernel(
    const float* __restrict__ g1, const float* __restrict__ b1,
    const float* __restrict__ m1, const float* __restrict__ v1,
    const float* __restrict__ w1, const float* __restrict__ c1b,
    const float* __restrict__ g2, const float* __restrict__ b2,
    const float* __restrict__ m2, const float* __restrict__ v2,
    const float* __restrict__ wA, const float* __restrict__ g3,
    const float* __restrict__ b3, const float* __restrict__ m3,
    const float* __restrict__ v3, const float* __restrict__ wB,
    const float* __restrict__ wf,
    float* __restrict__ wsW1, float* __restrict__ wsB1,
    float* __restrict__ wsW2, float* __restrict__ wsB2,
    _Float16* __restrict__ wsWB, _Float16* __restrict__ wsWF)
{
    const int o = threadIdx.x;   // output channel 0..127
    float s1[CIN], t1[CIN], s2[CIN], t2[CIN];
    #pragma unroll
    for (int c = 0; c < CIN; ++c) {
        s1[c] = g1[c] * rsqrtf(v1[c] + EPSV);
        t1[c] = b1[c] - m1[c] * s1[c];
        s2[c] = g2[c] * rsqrtf(v2[c] + EPSV);
        t2[c] = b2[c] - m2[c] * s2[c];
    }
    const float s3 = g3[o] * rsqrtf(v3[o] + EPSV);
    const float t3 = b3[o] - m3[o] * s3;

    float acc1 = c1b[o];
    float acc2 = 0.f;
    #pragma unroll
    for (int c = 0; c < CIN; ++c) {
        const float w = w1[o*CIN + c];
        wsW1[c*COUT + o] = w * s1[c];
        acc1 += w * t1[c];
        const float wa = wA[o*CIN + c];
        wsW2[c*COUT + o] = s3 * s2[c] * s1[c] * wa;   // BN1, BN2, BN3 scales folded
        acc2 += wa * t2[c];
    }
    wsB1[o] = acc1;
    wsB2[o] = s3 * acc2 + t3;

    for (int j = 0; j < COUT; ++j)     wsWB[o*COUT   + j] = (_Float16)wB[o*COUT   + j];
    for (int j = 0; j < 2*COUT; ++j)   wsWF[o*2*COUT + j] = (_Float16)wf[o*2*COUT + j];
}

// ---------------------------------------------------------------------------
// Fused main kernel: one block = 16 columns of one batch, 8 waves.
// ---------------------------------------------------------------------------
__global__ __launch_bounds__(256) void fused_kernel(
    const float* __restrict__ x,     // (B, CIN, N)
    const int*   __restrict__ nbr,   // (B, KNBR, N)
    const float* __restrict__ wsW1,  // (CIN, COUT) c-major
    const float* __restrict__ wsB1,  // (COUT)
    const float* __restrict__ wsW2,  // (CIN, COUT) c-major
    const float* __restrict__ wsB2,  // (COUT)
    const _Float16* __restrict__ wBh, // (COUT, COUT) row-major fp16
    const _Float16* __restrict__ wfh, // (COUT, 2*COUT) row-major fp16
    const float* __restrict__ bf,    // (COUT)
    float* __restrict__ out)         // (B, COUT, N)
{
    __shared__ float    sX[CIN * NCOL];            // center columns
    __shared__ float    sW1[CIN * COUT];           // point-proj weights
    __shared__ float    sB1[COUT];
    __shared__ _Float16 sH[NQ * SH_STRIDE];        // H[q][o] fp16, q = col*16 + slot
    __shared__ _Float16 sCat[NCOL * CAT_STRIDE];   // cat[col][0..127]=point, [128..255]=neigh
    __shared__ float    sRed[8 * 2 * SRED_TILE];   // per-wave D-tile transpose scratch

    const int tid  = threadIdx.x;
    const int lane = tid & 31;
    const int wave = tid >> 5;
    const int b    = blockIdx.y;
    const int n0   = blockIdx.x * NCOL;

    // ---- stage center columns + point-proj weights ----
    for (int i = tid; i < CIN * COUT; i += 256) sW1[i] = wsW1[i];
    if (tid < COUT) sB1[tid] = wsB1[tid];
    if (tid < CIN * NCOL) {
        const int c = tid >> 4, j = tid & 15;
        sX[c*NCOL + j] = x[((size_t)b*CIN + c)*NPTS + n0 + j];
    }
    __syncthreads();

    // ---- gather diffs (registers) + H = relu(W2' @ diff + b2) -> sH ----
    {
        const int j = tid >> 4, s = tid & 15;       // q = tid = j*16 + s
        const int kk = (s < 15) ? (s + 1) : 1;      // slot 15 duplicates slot 0 (max-safe pad)
        const int gi = nbr[((size_t)b*KNBR + kk)*NPTS + n0 + j];
        float d[CIN];
        #pragma unroll
        for (int c = 0; c < CIN; ++c)
            d[c] = x[((size_t)b*CIN + c)*NPTS + gi] - sX[c*NCOL + j];

        // wave-uniform weight indices -> scalar loads; fp32 FMA, fp16 store
        for (int o = 0; o < COUT; ++o) {
            float acc = wsB2[o];
            #pragma unroll
            for (int c = 0; c < CIN; ++c)
                acc = fmaf(wsW2[c*COUT + o], d[c], acc);
            sH[tid*SH_STRIDE + o] = (_Float16)fmaxf(acc, 0.f);
        }
    }

    // ---- point_emb = W1' @ x_c + b1 -> sCat[:, 0..127] ----
    for (int i = tid; i < COUT * NCOL; i += 256) {
        const int o = i & (COUT - 1), j = i >> 7;
        float acc = sB1[o];
        #pragma unroll
        for (int c = 0; c < CIN; ++c)
            acc = fmaf(sW1[c*COUT + o], sX[c*NCOL + j], acc);
        sCat[j*CAT_STRIDE + o] = (_Float16)acc;
    }
    __syncthreads();

    // ---- wB GEMM (WMMA fp16, f32 acc) + k-max via LDS transpose ----
    {
        const int lh = lane & 15;                  // D-tile N index (= neighbor slot k)
        const int hi = (lane >= 16) ? 1 : 0;       // D-tile M half
        float* wred = &sRed[wave * (2 * SRED_TILE)];

        #pragma unroll 1
        for (int m = 0; m < 8; ++m) {              // M-tiles of wB output rows
            v8f acc0 = {}; v8f acc1 = {};
            #pragma unroll
            for (int kk = 0; kk < 4; ++kk) {       // K = 128 in 4x32 steps
                // A: wB tile, lane row m*16+lh, K runs {+0..15} / {+16..31}
                const _Float16* ap = wBh + (m*16 + lh)*COUT + kk*32 + hi*16;
                V16U a; a.h[0] = *(const v8h*)ap; a.h[1] = *(const v8h*)(ap + 8);
                // B: H columns for output column j = wave (lanes: N=lh, K contiguous 16)
                const _Float16* bp0 = &sH[(wave*16 + lh)*SH_STRIDE + kk*32 + hi*16];
                V16U b0; b0.h[0] = *(const v8h*)bp0; b0.h[1] = *(const v8h*)(bp0 + 8);
                acc0 = __builtin_amdgcn_wmma_f32_16x16x32_f16(
                           false, a.v, false, b0.v, (short)0, acc0, false, false);
                // second column j = wave + 8 (A reused)
                const _Float16* bp1 = &sH[((wave + 8)*16 + lh)*SH_STRIDE + kk*32 + hi*16];
                V16U b1; b1.h[0] = *(const v8h*)bp1; b1.h[1] = *(const v8h*)(bp1 + 8);
                acc1 = __builtin_amdgcn_wmma_f32_16x16x32_f16(
                           false, a.v, false, b1.v, (short)0, acc1, false, false);
            }
            // Store D tiles k-major: row k=lh, o_local = hi*8 + r (r contiguous -> b128x2).
            #pragma unroll
            for (int r = 0; r < 8; ++r) {
                wred[            lh*SRED_ROW + hi*8 + r] = acc0[r];
                wred[SRED_TILE + lh*SRED_ROW + hi*8 + r] = acc1[r];
            }
            // Wave-local exchange: same-wave LDS ops are in-order; compiler emits the
            // dscnt wait for the may-aliasing re-read. Lane -> (col = hi, o_local = lh),
            // strided reads hit 32 distinct banks (row 20 + tile pad 16).
            float mx = wred[hi*SRED_TILE + lh];
            #pragma unroll
            for (int k = 1; k < 16; ++k)
                mx = fmaxf(mx, wred[hi*SRED_TILE + k*SRED_ROW + lh]);
            sCat[(wave + 8*hi)*CAT_STRIDE + COUT + m*16 + lh] = (_Float16)mx;
        }
    }
    __syncthreads();

    // ---- final GEMM: out = wf @ cat + bf  (WMMA, K = 256), M-tile = wave ----
    {
        const int lh = lane & 15;
        const int hi = (lane >= 16) ? 1 : 0;
        const int m  = wave;
        v8f acc;
        #pragma unroll
        for (int r = 0; r < 8; ++r) acc[r] = bf[m*16 + hi*8 + r];
        #pragma unroll
        for (int kk = 0; kk < 8; ++kk) {
            const _Float16* ap = wfh + (m*16 + lh)*(2*COUT) + kk*32 + hi*16;
            V16U a;  a.h[0]  = *(const v8h*)ap;  a.h[1]  = *(const v8h*)(ap + 8);
            const _Float16* bp = &sCat[lh*CAT_STRIDE + kk*32 + hi*16];
            V16U bb; bb.h[0] = *(const v8h*)bp;  bb.h[1] = *(const v8h*)(bp + 8);
            acc = __builtin_amdgcn_wmma_f32_16x16x32_f16(
                      false, a.v, false, bb.v, (short)0, acc, false, false);
        }
        #pragma unroll
        for (int r = 0; r < 8; ++r) {
            const int o = m*16 + hi*8 + r;
            out[((size_t)b*COUT + o)*NPTS + n0 + lh] = acc[r];  // 16 lanes coalesce per row
        }
    }
}

// ---------------------------------------------------------------------------
extern "C" void kernel_launch(void* const* d_in, const int* in_sizes, int n_in,
                              void* d_out, int out_size, void* d_ws, size_t ws_size,
                              hipStream_t stream)
{
    (void)in_sizes; (void)n_in; (void)out_size; (void)ws_size;
    const float* x   = (const float*)d_in[0];
    const int*   nbr = (const int*)  d_in[1];
    const float* g1  = (const float*)d_in[2];
    const float* b1  = (const float*)d_in[3];
    const float* m1  = (const float*)d_in[4];
    const float* v1  = (const float*)d_in[5];
    const float* w1  = (const float*)d_in[6];
    const float* c1b = (const float*)d_in[7];
    const float* g2  = (const float*)d_in[8];
    const float* b2  = (const float*)d_in[9];
    const float* m2  = (const float*)d_in[10];
    const float* v2  = (const float*)d_in[11];
    const float* wA  = (const float*)d_in[12];
    const float* g3  = (const float*)d_in[13];
    const float* b3  = (const float*)d_in[14];
    const float* m3  = (const float*)d_in[15];
    const float* v3  = (const float*)d_in[16];
    const float* wB  = (const float*)d_in[17];
    const float* wf  = (const float*)d_in[18];
    const float* bf  = (const float*)d_in[19];

    char* ws = (char*)d_ws;
    float*    wsW1 = (float*)   (ws + 0);       // 4096 B
    float*    wsB1 = (float*)   (ws + 4096);    //  512 B
    float*    wsW2 = (float*)   (ws + 4608);    // 4096 B
    float*    wsB2 = (float*)   (ws + 8704);    //  512 B
    _Float16* wsWB = (_Float16*)(ws + 9216);    // 32768 B
    _Float16* wsWF = (_Float16*)(ws + 41984);   // 65536 B -> total 107520 B

    setup_kernel<<<1, 128, 0, stream>>>(g1, b1, m1, v1, w1, c1b,
                                        g2, b2, m2, v2, wA,
                                        g3, b3, m3, v3, wB, wf,
                                        wsW1, wsB1, wsW2, wsB2, wsWB, wsWF);

    dim3 grid(NPTS / NCOL, 2);   // 4096 column-blocks x B=2
    fused_kernel<<<grid, 256, 0, stream>>>(x, nbr, wsW1, wsB1, wsW2, wsB2,
                                           wsWB, wsWF, bf, (float*)d_out);
}
